// SimCLR_Tree_18889266168353
// MI455X (gfx1250) — matrix operand order repaired and checked by reading, not compiled, not run on hardware
//
#include <hip/hip_runtime.h>
#include <math.h>

// Problem constants (BRANCH=16, DEPTH=3 balanced tree)
#define N1      1024
#define DPROJ   1024
#define MNODES  4368              // 4096 leaves + 256 depth-2 + 16 depth-1
#define N2      (N1 + MNODES)     // 5392
#define LEAVES  4096
#define INV_T   14.285714285714286f   // 1 / 0.07

#define NTILES_M 64               // 1024 / 16
#define NTILES_N 337              // 5392 / 16
#define NGROUPS  85               // ceil(337 / 4)  -- 4 N-tiles per wave

typedef __bf16 bf16;
typedef __attribute__((ext_vector_type(8)))  __bf16 v8bf;
typedef __attribute__((ext_vector_type(16))) __bf16 v16bf;
typedef __attribute__((ext_vector_type(8)))  float  v8f;

// ---------------------------------------------------------------------------
// Kernel 1: row-normalize q (rows 0..1023) and vitual_center (rows 1024..5391)
// into a single bf16 matrix kn[N2][DPROJ].  norm clamped at 1e-12 like ref.
// ---------------------------------------------------------------------------
__global__ void __launch_bounds__(256)
normalize_rows_kernel(const float* __restrict__ q,
                      const float* __restrict__ vc,
                      bf16* __restrict__ kn) {
    __shared__ float red[256];
    const int row = blockIdx.x;
    const float* src = (row < N1) ? (q + (size_t)row * DPROJ)
                                  : (vc + (size_t)(row - N1) * DPROJ);
    float x[4];
    float ss = 0.f;
#pragma unroll
    for (int u = 0; u < 4; ++u) {
        x[u] = src[threadIdx.x + u * 256];
        ss += x[u] * x[u];
    }
    red[threadIdx.x] = ss;
    __syncthreads();
    for (int off = 128; off > 0; off >>= 1) {
        if (threadIdx.x < off) red[threadIdx.x] += red[threadIdx.x + off];
        __syncthreads();
    }
    const float inv = 1.0f / fmaxf(sqrtf(red[0]), 1e-12f);
    bf16* dst = kn + (size_t)row * DPROJ;
#pragma unroll
    for (int u = 0; u < 4; ++u)
        dst[threadIdx.x + u * 256] = (bf16)(x[u] * inv);
}

// ---------------------------------------------------------------------------
// Kernel 2: logits[1024][5392] = (kn[0:1024] @ kn^T) * (1/T) via WMMA bf16.
// Register-blocked 1x4: each wave computes a 16x64 strip with 4 accumulators,
// loading its A fragment ONCE per K-step and reusing it across 4 WMMAs
// (2.5 b128 loads per v_wmma instead of 4).  The 8 waves of a block share the
// same N-group -> same 4 B tiles hot in WGP$/L0.
//
// Fragment layouts per CDNA5 ISA 7.12.2:
//   A (16x32 bf16): lane half h: a[0..7] = K (8h..8h+7), a[8..15] = K (16+8h..23+8h)
//   B (32x16 bf16): lane half h: b[0..7] = K (8h..8h+7), b[8..15] = K (16+8h..23+8h), row N = lane&15
//   C/D f32:        c[r] = (M = r + 8h, N = lane&15)
// ---------------------------------------------------------------------------
__global__ void __launch_bounds__(256)
wmma_gemm_logits_kernel(const bf16* __restrict__ kn,
                        float* __restrict__ logits) {
    const int lane = threadIdx.x & 31;
    const int wave = threadIdx.x >> 5;
    const int job  = blockIdx.x * 8 + wave;   // 64 * 85 = 5440 wave-jobs
    const int tm   = job & 63;                // M tile   (0..63)
    const int ng   = job >> 6;                // N group  (0..84), 4 tiles each
    const int half = lane >> 4;
    const int l15  = lane & 15;

    const bf16* __restrict__ arow = kn + (size_t)(tm * 16 + l15) * DPROJ;

    // B row pointers for the 4 N-tiles; clamp OOB tiles to the last valid
    // tile (duplicate compute, in-bounds reads; stores are guarded below).
    const bf16* brow[4];
#pragma unroll
    for (int t = 0; t < 4; ++t) {
        int tn = ng * 4 + t;
        if (tn > NTILES_N - 1) tn = NTILES_N - 1;
        brow[t] = kn + (size_t)(tn * 16 + l15) * DPROJ;
    }

    v8f acc0 = {}, acc1 = {}, acc2 = {}, acc3 = {};
#pragma unroll 2
    for (int k0 = 0; k0 < DPROJ; k0 += 32) {
        // A fragment: two 16B-aligned b128 loads, reused by all 4 WMMAs
        v8bf a0 = *(const v8bf*)(arow + k0 + 8 * half);
        v8bf a1 = *(const v8bf*)(arow + k0 + 16 + 8 * half);
        v16bf a = __builtin_shufflevector(a0, a1,
                      0,1,2,3,4,5,6,7,8,9,10,11,12,13,14,15);
#pragma unroll
        for (int t = 0; t < 4; ++t) {
            v8bf b0 = *(const v8bf*)(brow[t] + k0 + 8 * half);
            v8bf b1 = *(const v8bf*)(brow[t] + k0 + 16 + 8 * half);
            v16bf b = __builtin_shufflevector(b0, b1,
                          0,1,2,3,4,5,6,7,8,9,10,11,12,13,14,15);
            v8f c = (t == 0) ? acc0 : (t == 1) ? acc1 : (t == 2) ? acc2 : acc3;
            c = __builtin_amdgcn_wmma_f32_16x16x32_bf16(
                    /*neg_a=*/false, a, /*neg_b=*/false, b,
                    /*c_mod=*/(short)0, c, /*reuse_a=*/false, /*reuse_b=*/false);
            if      (t == 0) acc0 = c;
            else if (t == 1) acc1 = c;
            else if (t == 2) acc2 = c;
            else             acc3 = c;
        }
    }

    float* outrow = logits + (size_t)(tm * 16 + 8 * half) * N2 + l15;
#pragma unroll
    for (int t = 0; t < 4; ++t) {
        const int tn = ng * 4 + t;
        if (tn < NTILES_N) {                  // wave-uniform guard
            float* out = outrow + tn * 16;
            const v8f c = (t == 0) ? acc0 : (t == 1) ? acc1
                        : (t == 2) ? acc2 : acc3;
#pragma unroll
            for (int r = 0; r < 8; ++r)
                out[(size_t)r * N2] = c[r] * INV_T;
        }
    }
}

// ---------------------------------------------------------------------------
// Kernel 3: per-row masked positive-mean + masked logsumexp for the 3 tree
// depths, with pos/neg masks reconstructed analytically from leaf ids.
// full[nd] = [pos_logit, logits * nm]  (masked entries contribute exp(0)=1).
// |logits| <= 14.29 so direct expf sums stay well inside f32 range.
// ---------------------------------------------------------------------------
__global__ void __launch_bounds__(256)
masked_ce_rows_kernel(const float* __restrict__ logits,
                      const int* __restrict__ labels,
                      float* __restrict__ row_ce) {
    __shared__ float red[256];
    const int i  = blockIdx.x;
    const int ql = labels[i];       // query leaf id, 0..4095
    const int g2 = ql >> 4;         // depth-2 ancestor group
    const int g1 = ql >> 8;         // depth-1 ancestor group
    const float* __restrict__ lrow = logits + (size_t)i * N2;

    float ps0 = 0.f, ps1 = 0.f, ps2 = 0.f;   // pos sums
    float pc0 = 0.f, pc1 = 0.f, pc2 = 0.f;   // pos counts
    float es0 = 0.f, es1 = 0.f, es2 = 0.f;   // sum of exp over full[1:]

    for (int j = threadIdx.x; j < N2; j += 256) {
        const int n = (j < N1) ? labels[j] : (j - N1);   // node id of key j
        const float x = lrow[j];
        const bool isLeaf = n < LEAVES;
        const bool isD2   = (n >= LEAVES) && (n < LEAVES + 256);
        const bool isD1   = n >= LEAVES + 256;
        const bool inSub2 = (isD2 && (n - LEAVES) == g2) ||
                            (isLeaf && (n >> 4) == g2);
        const bool inSub1 = (isD1 && (n - (LEAVES + 256)) == g1) ||
                            (isD2 && ((n - LEAVES) >> 4) == g1) ||
                            (isLeaf && (n >> 8) == g1);
        const bool eq = (n == ql);
        const float ex = __expf(x);

        // nd=2 : pm = eq,               nm = !eq
        if (eq) { ps2 += x; pc2 += 1.f; }
        es2 += eq ? 1.f : ex;
        // nd=1 : pm = inSub2 && !eq,    nm = !inSub2
        if (inSub2 && !eq) { ps1 += x; pc1 += 1.f; }
        es1 += inSub2 ? 1.f : ex;
        // nd=0 : pm = inSub1 && !inSub2, nm = !inSub1
        if (inSub1 && !inSub2) { ps0 += x; pc0 += 1.f; }
        es0 += inSub1 ? 1.f : ex;
    }

    float vals[9] = {ps0, ps1, ps2, pc0, pc1, pc2, es0, es1, es2};
    float tot[9];
    for (int t = 0; t < 9; ++t) {
        red[threadIdx.x] = vals[t];
        __syncthreads();
        for (int off = 128; off > 0; off >>= 1) {
            if (threadIdx.x < off) red[threadIdx.x] += red[threadIdx.x + off];
            __syncthreads();
        }
        tot[t] = red[0];
        __syncthreads();
    }

    if (threadIdx.x == 0) {
        float ce = 0.f;
        for (int nd = 0; nd < 3; ++nd) {
            const float p = tot[nd] / fmaxf(tot[3 + nd], 1e-6f);
            ce += logf(__expf(p) + tot[6 + nd]) - p;   // lse(full) - full[0]
        }
        row_ce[i] = ce;   // mask==1, depth==3: ce/3 summed over 3 depths * 3
    }
}

// ---------------------------------------------------------------------------
// Kernel 4: deterministic final reduction.  loss = sum(row_ce) / N1.
// ---------------------------------------------------------------------------
__global__ void __launch_bounds__(256)
finalize_kernel(const float* __restrict__ row_ce, float* __restrict__ out) {
    __shared__ float red[256];
    float s = 0.f;
    for (int i = threadIdx.x; i < N1; i += 256) s += row_ce[i];
    red[threadIdx.x] = s;
    __syncthreads();
    for (int off = 128; off > 0; off >>= 1) {
        if (threadIdx.x < off) red[threadIdx.x] += red[threadIdx.x + off];
        __syncthreads();
    }
    if (threadIdx.x == 0) {
        const float loss = red[0] / (float)N1;
        out[0] = loss;   // reference returns (loss, loss)
        out[1] = loss;
    }
}

// ---------------------------------------------------------------------------
extern "C" void kernel_launch(void* const* d_in, const int* in_sizes, int n_in,
                              void* d_out, int out_size, void* d_ws, size_t ws_size,
                              hipStream_t stream) {
    const float* q      = (const float*)d_in[0];   // [1024,1024]
    const float* vc     = (const float*)d_in[1];   // [4368,1024]
    const int*   labels = (const int*)d_in[2];     // [1024]
    // d_in[3..6] (pos/neg/mask/depth) are constant tree structure -> folded.

    // workspace layout (all offsets 256B aligned):
    //   kn     : bf16 [5392][1024]  = 11,042,816 B
    //   logits : f32  [1024][5392]  = 22,085,632 B
    //   row_ce : f32  [1024]        =      4,096 B
    bf16*  kn     = (bf16*)d_ws;
    float* logits = (float*)((char*)d_ws + (size_t)N2 * DPROJ * sizeof(bf16));
    float* row_ce = logits + (size_t)N1 * N2;

    normalize_rows_kernel<<<N2, 256, 0, stream>>>(q, vc, kn);
    wmma_gemm_logits_kernel<<<(NTILES_M * NGROUPS) / 8, 256, 0, stream>>>(kn, logits);
    masked_ce_rows_kernel<<<N1, 256, 0, stream>>>(logits, labels, row_ce);
    finalize_kernel<<<1, 256, 0, stream>>>(row_ce, (float*)d_out);
}
